// AttentionalSplatting_42348377539177
// MI455X (gfx1250) — matrix operand order repaired and checked by reading, not compiled, not run on hardware
//
#include <hip/hip_runtime.h>
#include <hip/hip_bf16.h>
#include <math.h>

// ---------------- problem constants ----------------
constexpr int kT  = 4;
constexpr int kM  = 1024;
constexpr int kHW = 2304;
constexpr int kDF = 256;
constexpr int kH  = 8;
constexpr int kDK = 32;                  // d_k per head
constexpr float kEPS   = 1e-6f;
constexpr float kSCALE = 0.17677669529663687f;   // 1/sqrt(32)
constexpr int kQT = kHW / 16;            // 144 query tiles per t

// ---------------- workspace layout (offsets in _Float16 units) ----------------
// Weights are stored TRANSPOSED ([n][k] row-major) so B-fragments are contiguous.
constexpr size_t WS_WQ  = 0;                       // Wq^T   256*256
constexpr size_t WS_WK  = WS_WQ  + 65536;          // Wk^T
constexpr size_t WS_WV  = WS_WK  + 65536;          // Wv^T
constexpr size_t WS_WC  = WS_WV  + 65536;          // (Wo @ W_out_w)^T
constexpr size_t WS_K   = WS_WC  + 65536;          // [t][h][m][dk], L2-normed
constexpr size_t WS_VT  = WS_K   + (size_t)kT*kH*kM*kDK;   // [t][h][dk][m]
constexpr size_t WS_Q   = WS_VT  + (size_t)kT*kH*kDK*kM;   // [t][h][q][dk], L2-normed
constexpr size_t WS_ATT = WS_Q   + (size_t)kT*kH*kHW*kDK;  // [t][q][256]

typedef __attribute__((ext_vector_type(16))) _Float16 v16h;
typedef __attribute__((ext_vector_type(8)))  _Float16 v8h;
typedef __attribute__((ext_vector_type(4)))  _Float16 v4h;
typedef __attribute__((ext_vector_type(8)))  float    v8f;

__device__ __forceinline__ v8f wmma_f16(v16h a, v16h b, v8f c) {
  // D = A(16x32 f16) * B(32x16 f16) + C(16x16 f32)
  return __builtin_amdgcn_wmma_f32_16x16x32_f16(false, a, false, b, (short)0, c,
                                                false, false);
}

__device__ __forceinline__ v8f vzero8() {
  v8f c;
#pragma unroll
  for (int r = 0; r < 8; ++r) c[r] = 0.0f;
  return c;
}

__device__ __forceinline__ v16h cat8(v8h lo, v8h hi) {
  return __builtin_shufflevector(lo, hi, 0, 1, 2, 3, 4, 5, 6, 7,
                                          8, 9, 10, 11, 12, 13, 14, 15);
}

// A-fragment (16x32 f16, ISA §7.12.2): lane holds row (lane&15);
// elements 0..7 -> K = kbase + (lane>=16?8:0) + 0..7, elements 8..15 -> +16.
// Two 16-byte loads per lane.
__device__ __forceinline__ v16h load_afrag(const _Float16* rowptr, int kbase, int lane) {
  const v8h* p = (const v8h*)(rowptr + kbase + ((lane >> 4) << 3));
  return cat8(p[0], p[2]);     // +0 halfs and +16 halfs
}

// B-fragment (32x16 f16): lane holds column (lane&15); same K order as A:
// elements 0..7 -> K = (lane>=16?8:0) + 0..7, elements 8..15 -> +16. `p` points at B^T row n
// (k contiguous in memory), i.e. p[k] == B[k][n]. Two 16-byte loads per lane.
__device__ __forceinline__ v16h load_bfrag(const _Float16* p, int lane) {
  const v8h* q = (const v8h*)(p + ((lane >> 4) << 3));
  return cat8(q[0], q[2]);
}

// ---------------- kernel 1: weight prep ----------------
// Store f16 TRANSPOSED weights; Wcomb = Wo @ W_out_w folds the two output GEMMs.
__global__ void __launch_bounds__(256)
prep_kernel(const float* __restrict__ Wq, const float* __restrict__ Wk,
            const float* __restrict__ Wv, const float* __restrict__ Wo,
            const float* __restrict__ Wow, _Float16* __restrict__ ws) {
  int k = blockIdx.x;      // input-dim index (row of W)
  int n = threadIdx.x;     // output-dim index (col of W)
  size_t src = (size_t)k * kDF + n;
  size_t dst = (size_t)n * kDF + k;      // transposed
  ws[WS_WQ + dst] = (_Float16)Wq[src];
  ws[WS_WK + dst] = (_Float16)Wk[src];
  ws[WS_WV + dst] = (_Float16)Wv[src];
  // Wcomb[k][n] = sum_c Wo[k][c] * Wow[c][n]; store transposed at [n][k]
  float acc = 0.0f;
  for (int c = 0; c < kDF; ++c) acc += Wo[(size_t)k * kDF + c] * Wow[(size_t)c * kDF + n];
  ws[WS_WC + dst] = (_Float16)acc;
}

// ---------------- kernel 2: K/V projection (+ L2 norm of K, V transposed) ----
__global__ void __launch_bounds__(256)
kv_kernel(const float* __restrict__ tokens, _Float16* __restrict__ ws) {
  __shared__ _Float16 sA[16][kDF];     // token tile f16
  __shared__ float sK[8][16][kDK];
  __shared__ float sV[8][16][kDK];
  int t = blockIdx.x >> 6;             // 64 m-tiles per t
  int m0 = (blockIdx.x & 63) * 16;

  const float4* src4 = (const float4*)(tokens + ((size_t)t * kM + m0) * kDF);
  v4h* sA4 = (v4h*)(&sA[0][0]);
  for (int i = threadIdx.x; i < 16 * kDF / 4; i += 256) {
    float4 f = src4[i];
    v4h h; h[0] = (_Float16)f.x; h[1] = (_Float16)f.y;
           h[2] = (_Float16)f.z; h[3] = (_Float16)f.w;
    sA4[i] = h;
  }
  __syncthreads();

  int wave = threadIdx.x >> 5, lane = threadIdx.x & 31;
  int h = wave;                                   // one head per wave
  int n0 = h * kDK + (lane & 15);
  const _Float16* wk0 = ws + WS_WK + (size_t)n0 * kDF;
  const _Float16* wk1 = ws + WS_WK + (size_t)(n0 + 16) * kDF;
  const _Float16* wv0 = ws + WS_WV + (size_t)n0 * kDF;
  const _Float16* wv1 = ws + WS_WV + (size_t)(n0 + 16) * kDF;

  v8f k0 = vzero8(), k1 = vzero8(), v0 = vzero8(), v1 = vzero8();
  for (int kk = 0; kk < 8; ++kk) {
    v16h a = load_afrag(&sA[lane & 15][0], kk * 32, lane);
    k0 = wmma_f16(a, load_bfrag(wk0 + kk * 32, lane), k0);
    k1 = wmma_f16(a, load_bfrag(wk1 + kk * 32, lane), k1);
    v0 = wmma_f16(a, load_bfrag(wv0 + kk * 32, lane), v0);
    v1 = wmma_f16(a, load_bfrag(wv1 + kk * 32, lane), v1);
  }
#pragma unroll
  for (int r = 0; r < 8; ++r) {
    int row = r + ((lane >> 4) << 3);
    int col = lane & 15;
    sK[wave][row][col]      = k0[r];  sK[wave][row][col + 16] = k1[r];
    sV[wave][row][col]      = v0[r];  sV[wave][row][col + 16] = v1[r];
  }
  __syncthreads();

  // K: per-row L2 norm, store [t][h][m][dk] with 16B vector stores
  if (lane < 16) {
    int row = lane;
    float ss = 0.0f;
    for (int d = 0; d < kDK; ++d) { float x = sK[wave][row][d]; ss += x * x; }
    float inv = 1.0f / (sqrtf(ss) + kEPS);
    v8h* dst = (v8h*)(ws + WS_K + ((size_t)(t * kH + h) * kM + m0 + row) * kDK);
#pragma unroll
    for (int d8 = 0; d8 < 4; ++d8) {
      v8h hv;
#pragma unroll
      for (int d = 0; d < 8; ++d) hv[d] = (_Float16)(sK[wave][row][d8 * 8 + d] * inv);
      dst[d8] = hv;
    }
  }
  // V: store transposed [t][h][dk][m] so PV B-fragments are contiguous
  {
    int d = lane;
    _Float16* dst = ws + WS_VT + ((size_t)(t * kH + h) * kDK + d) * kM + m0;
    for (int row = 0; row < 16; ++row) dst[row] = (_Float16)sV[wave][row][d];
  }
}

// ---------------- kernel 3: Q generation (coord proj + 2D RoPE + Wq + norm) --
__global__ void __launch_bounds__(256)
q_kernel(const float* __restrict__ pos, const float* __restrict__ Wc,
         const float* __restrict__ bc, _Float16* __restrict__ ws) {
  __shared__ _Float16 sA[16][kDF];
  __shared__ float sC[8][16][kDK];
  int t  = blockIdx.x / kQT;
  int q0 = (blockIdx.x % kQT) * 16;

  // Q_in = pos @ W_coord + b, then 2D RoPE; write f16 tile into LDS
  for (int p = threadIdx.x; p < 16 * 128; p += 256) {
    int row = p >> 7;
    int pj  = p & 127;
    int hs  = pj >> 6;                 // 0: x-rotated half, 1: y-rotated half
    int i   = pj & 63;
    int j0  = hs * 128 + 2 * i;
    const float* pp = pos + ((size_t)t * kHW + q0 + row) * 2;
    float px = pp[0], py = pp[1];
    float f0 = px * Wc[j0]     + py * Wc[kDF + j0]     + bc[j0];
    float f1 = px * Wc[j0 + 1] + py * Wc[kDF + j0 + 1] + bc[j0 + 1];
    float theta = expf(-(float)i * (4.605170185988091f / 64.0f)); // 100^(-i/64)
    float ang = (hs ? py : px) * theta;
    float c = cosf(ang), s = sinf(ang);
    sA[row][j0]     = (_Float16)(f0 * c - f1 * s);
    sA[row][j0 + 1] = (_Float16)(f0 * s + f1 * c);
  }
  __syncthreads();

  int wave = threadIdx.x >> 5, lane = threadIdx.x & 31;
  int h = wave;
  int n0 = h * kDK + (lane & 15);
  const _Float16* wq0 = ws + WS_WQ + (size_t)n0 * kDF;
  const _Float16* wq1 = ws + WS_WQ + (size_t)(n0 + 16) * kDF;
  v8f c0 = vzero8(), c1 = vzero8();
  for (int kk = 0; kk < 8; ++kk) {
    v16h a = load_afrag(&sA[lane & 15][0], kk * 32, lane);
    c0 = wmma_f16(a, load_bfrag(wq0 + kk * 32, lane), c0);
    c1 = wmma_f16(a, load_bfrag(wq1 + kk * 32, lane), c1);
  }
#pragma unroll
  for (int r = 0; r < 8; ++r) {
    int row = r + ((lane >> 4) << 3);
    int col = lane & 15;
    sC[wave][row][col]      = c0[r];
    sC[wave][row][col + 16] = c1[r];
  }
  __syncthreads();
  if (lane < 16) {
    int row = lane;
    float ss = 0.0f;
    for (int d = 0; d < kDK; ++d) { float x = sC[wave][row][d]; ss += x * x; }
    float inv = 1.0f / (sqrtf(ss) + kEPS);
    v8h* dst = (v8h*)(ws + WS_Q + ((size_t)(t * kH + h) * kHW + q0 + row) * kDK);
#pragma unroll
    for (int d8 = 0; d8 < 4; ++d8) {
      v8h hv;
#pragma unroll
      for (int d = 0; d < 8; ++d) hv[d] = (_Float16)(sC[wave][row][d8 * 8 + d] * inv);
      dst[d8] = hv;
    }
  }
}

// ---------------- kernel 4: fused attention per (t, h, 16-query tile) --------
__global__ void __launch_bounds__(256)
att_kernel(const float* __restrict__ sbias, _Float16* __restrict__ ws) {
  __shared__ _Float16 sS[16][kM];      // scores -> P (f16, 32 KB)
  __shared__ float sRed[16][16];
  __shared__ float sMax[16];
  __shared__ float sInv[16];
  __shared__ float sPart[8][16][kDK];

  int t   = blockIdx.x / (kH * kQT);
  int rem = blockIdx.x % (kH * kQT);
  int h   = rem / kQT;
  int q0  = (rem % kQT) * 16;
  int wave = threadIdx.x >> 5, lane = threadIdx.x & 31;

  const _Float16* QH  = ws + WS_Q  + (size_t)(t * kH + h) * kHW * kDK;
  const _Float16* KH  = ws + WS_K  + (size_t)(t * kH + h) * kM  * kDK;
  const _Float16* VtH = ws + WS_VT + (size_t)(t * kH + h) * kDK * kM;

  // Q fragment: full dk=32 == one WMMA K-step
  v16h aQ = load_afrag(QH + (size_t)(q0 + (lane & 15)) * kDK, 0, lane);

  // ---- phase 1: scores (each wave: 8 of the 64 key tiles) ----
  for (int kt = wave * 8; kt < wave * 8 + 8; ++kt) {
    int mcol = kt * 16 + (lane & 15);
    v16h bK = load_bfrag(KH + (size_t)mcol * kDK, lane);   // K^T column
    v8f c = vzero8();
    c = wmma_f16(aQ, bK, c);
#pragma unroll
    for (int r = 0; r < 8; ++r) {
      int row = r + ((lane >> 4) << 3);
      float b = sbias[((size_t)t * kM + mcol) * kHW + q0 + row];
      sS[row][mcol] = (_Float16)(c[r] * kSCALE + b);
    }
  }
  __syncthreads();

  // ---- phase 2: softmax over M=1024 (unnormalized P, fold 1/sum later) ----
  int row = threadIdx.x & 15, seg = threadIdx.x >> 4;   // 16 segments of 64
  {
    float mx = -1e30f;
    for (int j = seg * 64; j < seg * 64 + 64; ++j)
      mx = fmaxf(mx, (float)sS[row][j]);
    sRed[row][seg] = mx;
  }
  __syncthreads();
  if (threadIdx.x < 16) {
    float mx = -1e30f;
    for (int s = 0; s < 16; ++s) mx = fmaxf(mx, sRed[threadIdx.x][s]);
    sMax[threadIdx.x] = mx;
  }
  __syncthreads();
  {
    float mx = sMax[row];
    float sum = 0.0f;
    for (int j = seg * 64; j < seg * 64 + 64; ++j) {
      float p = expf((float)sS[row][j] - mx);
      sS[row][j] = (_Float16)p;          // in-place, same thread
      sum += p;
    }
    sRed[row][seg] = sum;
  }
  __syncthreads();
  if (threadIdx.x < 16) {
    float sum = 0.0f;
    for (int s = 0; s < 16; ++s) sum += sRed[threadIdx.x][s];
    sInv[threadIdx.x] = 1.0f / sum;
  }
  __syncthreads();

  // ---- phase 3: P @ V (each wave: 4 of the 32 m-chunks of K=32) ----
  v8f o0 = vzero8(), o1 = vzero8();
  for (int mc = wave; mc < 32; mc += 8) {
    v16h aP  = load_afrag(&sS[lane & 15][0], mc * 32, lane);
    v16h bV0 = load_bfrag(VtH + (size_t)(lane & 15)        * kM + mc * 32, lane);
    v16h bV1 = load_bfrag(VtH + (size_t)(16 + (lane & 15)) * kM + mc * 32, lane);
    o0 = wmma_f16(aP, bV0, o0);
    o1 = wmma_f16(aP, bV1, o1);
  }
#pragma unroll
  for (int r = 0; r < 8; ++r) {
    int rr = r + ((lane >> 4) << 3);
    sPart[wave][rr][lane & 15]        = o0[r];
    sPart[wave][rr][16 + (lane & 15)] = o1[r];
  }
  __syncthreads();
  for (int j = threadIdx.x; j < 16 * kDK; j += 256) {
    int rr = j >> 5, col = j & 31;
    float acc = 0.0f;
    for (int w = 0; w < 8; ++w) acc += sPart[w][rr][col];
    acc *= sInv[rr];
    ws[WS_ATT + (size_t)(t * kHW + q0 + rr) * kDF + h * kDK + col] = (_Float16)acc;
  }
}

// ---------------- kernel 5: attended @ Wcomb + bias + residual ----------------
__global__ void __launch_bounds__(256)
out_kernel(const float* __restrict__ fmap, const float* __restrict__ bout,
           const _Float16* __restrict__ ws, float* __restrict__ out) {
  int r0 = blockIdx.x * 16;                       // global row over T*HW
  int wave = threadIdx.x >> 5, lane = threadIdx.x & 31;
  const _Float16* A = ws + WS_ATT;

  v8f c0 = vzero8(), c1 = vzero8();
  const _Float16* arow = A + (size_t)(r0 + (lane & 15)) * kDF;
  int n0 = wave * 32 + (lane & 15);
  const _Float16* w0 = ws + WS_WC + (size_t)n0 * kDF;
  const _Float16* w1 = ws + WS_WC + (size_t)(n0 + 16) * kDF;
  for (int kk = 0; kk < 8; ++kk) {
    v16h a = load_afrag(arow, kk * 32, lane);
    c0 = wmma_f16(a, load_bfrag(w0 + kk * 32, lane), c0);
    c1 = wmma_f16(a, load_bfrag(w1 + kk * 32, lane), c1);
  }
#pragma unroll
  for (int r = 0; r < 8; ++r) {
    int row  = r0 + r + ((lane >> 4) << 3);
    int col0 = wave * 32 + (lane & 15);
    int col1 = col0 + 16;
    size_t i0 = (size_t)row * kDF + col0;
    size_t i1 = (size_t)row * kDF + col1;
    out[i0] = c0[r] + bout[col0] + fmap[i0];
    out[i1] = c1[r] + bout[col1] + fmap[i1];
  }
}

// ---------------- host launcher ----------------
extern "C" void kernel_launch(void* const* d_in, const int* in_sizes, int n_in,
                              void* d_out, int out_size, void* d_ws, size_t ws_size,
                              hipStream_t stream) {
  const float* track_tokens = (const float*)d_in[0];
  const float* feature_map  = (const float*)d_in[1];
  const float* feature_pos  = (const float*)d_in[2];
  const float* spatial_bias = (const float*)d_in[3];
  const float* Wq           = (const float*)d_in[4];
  const float* Wk           = (const float*)d_in[5];
  const float* Wv           = (const float*)d_in[6];
  const float* Wo           = (const float*)d_in[7];
  const float* W_out_w      = (const float*)d_in[8];
  const float* W_out_b      = (const float*)d_in[9];
  const float* W_coord_w    = (const float*)d_in[10];
  const float* W_coord_b    = (const float*)d_in[11];
  float* out = (float*)d_out;
  _Float16* ws = (_Float16*)d_ws;

  prep_kernel<<<256, 256, 0, stream>>>(Wq, Wk, Wv, Wo, W_out_w, ws);
  kv_kernel<<<kT * kM / 16, 256, 0, stream>>>(track_tokens, ws);
  q_kernel<<<kT * kQT, 256, 0, stream>>>(feature_pos, W_coord_w, W_coord_b, ws);
  att_kernel<<<kT * kH * kQT, 256, 0, stream>>>(spatial_bias, ws);
  out_kernel<<<kT * kHW / 16, 256, 0, stream>>>(feature_map, W_out_b, ws, out);
}
